// RGNN_53755810677120
// MI455X (gfx1250) — matrix operand order repaired and checked by reading, not compiled, not run on hardware
//
#include <hip/hip_runtime.h>
#include <hip/hip_bf16.h>

typedef __attribute__((ext_vector_type(16))) _Float16 v16h;
typedef __attribute__((ext_vector_type(8)))  float    v8f;

#define NHID 32
#define SEQ  4
#define FINAL 128
#define NG   64

// ---------------------------------------------------------------------------
// Kernel 1: encoder. pad chunks [16,32,32,32] -> [4,32], add sinusoidal PE,
// per-token linear 32->32.  One node per 128-thread block; thread = (s,d).
// ---------------------------------------------------------------------------
__global__ void __launch_bounds__(128)
encode_kernel(const float* __restrict__ x, const float* __restrict__ W,
              float* __restrict__ h, int N)
{
    __shared__ float sW[32 * 32];
    __shared__ float sx[SEQ][32];
    int t = threadIdx.x;
    int n = blockIdx.x;
    if (n >= N) return;

    for (int i = t; i < 32 * 32; i += 128) sW[i] = W[i];

    int s = t >> 5, k = t & 31;
    // positional encoding pe[s][k]
    int   i2  = k >> 1;
    float div = __expf(-(float)(2 * i2) * 0.2878231366242558f); // ln(10000)/32
    float ang = (float)s * div;
    float pe  = (k & 1) ? __cosf(ang) : __sinf(ang);
    // padded input chunk
    int   start = (s == 0) ? 0 : (16 + 32 * (s - 1));
    int   len   = (s == 0) ? 16 : 32;
    float xv    = (k < len) ? x[(size_t)n * 112 + start + k] : 0.0f;
    sx[s][k] = xv + pe;
    __syncthreads();

    int dcol = t & 31;
    float acc = 0.0f;
#pragma unroll
    for (int kk = 0; kk < 32; ++kk) acc += sx[s][kk] * sW[kk * 32 + dcol];
    h[(size_t)n * FINAL + t] = acc;
}

// ---------------------------------------------------------------------------
// Kernel 2: feature attention, WMMA f16.  4 waves/block, 4 nodes/wave
// (16 token rows of 32 features).  10 WMMAs per wave:
//   QKV (6 tiles of 16x16x32), FFN1 (2), FFN2 (2).
// A-fragment layout (ISA 7.12.2, 16-bit A 16x32): lane L holds row M=L&15,
//   halves 0..7 -> K = kbA..kbA+7, halves 8..15 -> K = kbA+16..kbA+23,
//   kbA = (L<16) ? 0 : 8.
// B-fragment (32x16): lane L col=L&15, elements in the same K order as A above
// D (f32 16x16): VGPR v -> row v + 8*(lane>>4), col = lane&15.
// ---------------------------------------------------------------------------
__global__ void __launch_bounds__(128)
featatt_kernel(const float* __restrict__ h,
               const float* __restrict__ Wqkv,  // [32][96]
               const float* __restrict__ W1,    // [32][32]
               const float* __restrict__ B1,    // [32]
               const float* __restrict__ W2,    // [32][32]
               const float* __restrict__ B2,    // [32]
               float* __restrict__ out, int N)
{
    __shared__ _Float16 s_wqkv[32 * 96];
    __shared__ _Float16 s_w1[32 * 32];
    __shared__ _Float16 s_w2[32 * 32];
    __shared__ float    s_b1[32], s_b2[32];
    __shared__ float    s_buf[4][16 * 96];  // per wave: cols 0-31 Q->y, 32-63 K->z, 64-95 V
    __shared__ float    s_att[4][16 * 4];

    int t    = threadIdx.x;
    int wave = t >> 5;
    int lane = t & 31;

    for (int i = t; i < 32 * 96; i += 128) s_wqkv[i] = (_Float16)Wqkv[i];
    for (int i = t; i < 32 * 32; i += 128) {
        s_w1[i] = (_Float16)W1[i];
        s_w2[i] = (_Float16)W2[i];
    }
    if (t < 32) { s_b1[t] = B1[t]; s_b2[t] = B2[t]; }
    __syncthreads();

    int M    = lane & 15;
    int hi   = lane >> 4;
    int kbA  = hi ? 8 : 0;    // A fragment K base
    int kbB  = hi ? 16 : 0;   // unused; B loads use kbA
    int colL = lane & 15;
    int n0   = (blockIdx.x * 4 + wave) * 4;  // first node of this wave

    // ---- load A fragment: 16 token rows x 32 feats (f32 -> f16) ----
    int nodeL = n0 + (M >> 2);
    int nodeC = nodeL < N ? nodeL : (N - 1);
    const float* hrow = h + (size_t)nodeC * FINAL + (M & 3) * 32;
    v16h a;
#pragma unroll
    for (int i = 0; i < 8; ++i) {
        a[i]     = (_Float16)hrow[kbA + i];
        a[8 + i] = (_Float16)hrow[kbA + 16 + i];
    }

    // ---- QKV: 6 WMMA tiles over 96 output columns ----
    float* buf = s_buf[wave];
#pragma unroll
    for (int c = 0; c < 6; ++c) {
        int col = colL + 16 * c;
        v16h b;
#pragma unroll
        for (int i = 0; i < 16; ++i) b[i] = s_wqkv[(kbA + i + (i >= 8 ? 8 : 0)) * 96 + col];
        v8f z = {};
        v8f d = __builtin_amdgcn_wmma_f32_16x16x32_f16(false, a, false, b,
                                                       (short)0, z, false, false);
#pragma unroll
        for (int v = 0; v < 8; ++v) buf[(v + 8 * hi) * 96 + col] = d[v];
    }
    __syncthreads();

    // ---- softmax attention over 4 tokens (lanes 0..15, one row each) ----
    if (lane < 16) {
        int base = M & ~3;
        float sc[4];
#pragma unroll
        for (int u = 0; u < 4; ++u) {
            float s = 0.0f;
#pragma unroll
            for (int dd = 0; dd < 32; ++dd)
                s += buf[M * 96 + dd] * buf[(base + u) * 96 + 32 + dd];
            sc[u] = s * 0.17677669529663687f; // 1/sqrt(32)
        }
        float mx = fmaxf(fmaxf(sc[0], sc[1]), fmaxf(sc[2], sc[3]));
        float e0 = __expf(sc[0] - mx), e1 = __expf(sc[1] - mx);
        float e2 = __expf(sc[2] - mx), e3 = __expf(sc[3] - mx);
        float inv = 1.0f / (e0 + e1 + e2 + e3);
        s_att[wave][M * 4 + 0] = e0 * inv;
        s_att[wave][M * 4 + 1] = e1 * inv;
        s_att[wave][M * 4 + 2] = e2 * inv;
        s_att[wave][M * 4 + 3] = e3 * inv;
    }
    __syncthreads();

    // ---- y = att @ V, built directly in A-fragment layout; y also to LDS ----
    int base = M & ~3;
    float au0 = s_att[wave][M * 4 + 0], au1 = s_att[wave][M * 4 + 1];
    float au2 = s_att[wave][M * 4 + 2], au3 = s_att[wave][M * 4 + 3];
    v16h a2;
#pragma unroll
    for (int i = 0; i < 16; ++i) {
        int dd = (i < 8) ? (kbA + i) : (kbA + 16 + (i - 8));
        float yv = au0 * buf[(base + 0) * 96 + 64 + dd]
                 + au1 * buf[(base + 1) * 96 + 64 + dd]
                 + au2 * buf[(base + 2) * 96 + 64 + dd]
                 + au3 * buf[(base + 3) * 96 + 64 + dd];
        buf[M * 96 + dd] = yv;          // y into (retired) Q region
        a2[i] = (_Float16)yv;
    }

    // ---- FFN1: z = relu(y @ W1 + b1) -> LDS (retired K region) ----
#pragma unroll
    for (int c = 0; c < 2; ++c) {
        int col = colL + 16 * c;
        v16h b;
#pragma unroll
        for (int i = 0; i < 16; ++i) b[i] = s_w1[(kbA + i + (i >= 8 ? 8 : 0)) * 32 + col];
        v8f z = {};
        v8f d = __builtin_amdgcn_wmma_f32_16x16x32_f16(false, a2, false, b,
                                                       (short)0, z, false, false);
#pragma unroll
        for (int v = 0; v < 8; ++v) {
            int r = v + 8 * hi;
            buf[r * 96 + 32 + col] = fmaxf(d[v] + s_b1[col], 0.0f);
        }
    }
    __syncthreads();

    // ---- gather z as A fragment ----
    v16h a3;
#pragma unroll
    for (int i = 0; i < 8; ++i) {
        a3[i]     = (_Float16)buf[M * 96 + 32 + kbA + i];
        a3[8 + i] = (_Float16)buf[M * 96 + 32 + kbA + 16 + i];
    }

    // ---- FFN2 + residual + store ----
#pragma unroll
    for (int c = 0; c < 2; ++c) {
        int col = colL + 16 * c;
        v16h b;
#pragma unroll
        for (int i = 0; i < 16; ++i) b[i] = s_w2[(kbA + i + (i >= 8 ? 8 : 0)) * 32 + col];
        v8f z = {};
        v8f d = __builtin_amdgcn_wmma_f32_16x16x32_f16(false, a3, false, b,
                                                       (short)0, z, false, false);
#pragma unroll
        for (int v = 0; v < 8; ++v) {
            int r  = v + 8 * hi;
            int nn = n0 + (r >> 2);
            if (nn < N) {
                float val = buf[r * 96 + col] + d[v] + s_b2[col];
                out[(size_t)nn * FINAL + (r & 3) * 32 + col] = val;
            }
        }
    }
}

// ---------------------------------------------------------------------------
// Kernel 3: edge scatter-add.  One wave per edge, float4 gather, 4 atomic
// f32 adds per lane (128 floats/edge).
// ---------------------------------------------------------------------------
__global__ void __launch_bounds__(256)
scatter_kernel(const float* __restrict__ xt, const int* __restrict__ src,
               const int* __restrict__ dst, int E, float* __restrict__ acc)
{
    int t = blockIdx.x * 256 + threadIdx.x;
    int e = t >> 5;
    if (e >= E) return;
    int lane = t & 31;
    int s  = src[e];
    int d0 = dst[e];
    const float4* sp = (const float4*)(xt + (size_t)s * FINAL);
    float4 v = sp[lane];
    float* dp = acc + (size_t)d0 * FINAL + lane * 4;
    unsafeAtomicAdd(dp + 0, v.x);
    unsafeAtomicAdd(dp + 1, v.y);
    unsafeAtomicAdd(dp + 2, v.z);
    unsafeAtomicAdd(dp + 3, v.w);
}

// ---------------------------------------------------------------------------
// Kernel 4: elementwise ReLU (float4), xacc -> h.
// ---------------------------------------------------------------------------
__global__ void __launch_bounds__(256)
relu_kernel(const float4* __restrict__ in, float4* __restrict__ out, int n4)
{
    int i = blockIdx.x * 256 + threadIdx.x;
    if (i < n4) {
        float4 v = in[i];
        v.x = fmaxf(v.x, 0.0f); v.y = fmaxf(v.y, 0.0f);
        v.z = fmaxf(v.z, 0.0f); v.w = fmaxf(v.w, 0.0f);
        out[i] = v;
    }
}

// ---------------------------------------------------------------------------
// Kernel 5: segment mean-pool stage 1.  batch[] is sorted, so each block
// run-length compresses before issuing atomics.  128 threads = 1 feature
// column each, 256 nodes per block.
// ---------------------------------------------------------------------------
__global__ void __launch_bounds__(128)
pool_kernel(const float* __restrict__ h, const int* __restrict__ batch, int N,
            float* __restrict__ sums, float* __restrict__ cnts)
{
    const int NPB = 256;
    int f  = threadIdx.x;
    int n0 = blockIdx.x * NPB;
    if (n0 >= N) return;
    int nEnd = n0 + NPB; if (nEnd > N) nEnd = N;

    int   cur = batch[n0];
    float acc = 0.0f;
    int   run = 0;
    for (int n = n0; n < nEnd; ++n) {
        int b = batch[n];
        if (b != cur) {
            unsafeAtomicAdd(&sums[cur * FINAL + f], acc);
            if (f == 0) unsafeAtomicAdd(&cnts[cur], (float)run);
            cur = b; acc = 0.0f; run = 0;
        }
        acc += h[(size_t)n * FINAL + f];
        run++;
    }
    unsafeAtomicAdd(&sums[cur * FINAL + f], acc);
    if (f == 0) unsafeAtomicAdd(&cnts[cur], (float)run);
}

// ---------------------------------------------------------------------------
// Kernel 6: g = sums/cnt; out = relu(g@W1+b1)@W2+b2.  One block.
// ---------------------------------------------------------------------------
__global__ void __launch_bounds__(128)
final_kernel(const float* __restrict__ sums, const float* __restrict__ cnts,
             const float* __restrict__ w1, const float* __restrict__ b1,
             const float* __restrict__ w2, const float* __restrict__ b2,
             float* __restrict__ out)
{
    __shared__ float g[NG * FINAL];
    __shared__ float hdn[NG * FINAL];
    int t = threadIdx.x;
    for (int i = t; i < NG * FINAL; i += 128) {
        int gi = i >> 7;
        g[i] = sums[i] / fmaxf(cnts[gi], 1.0f);
    }
    __syncthreads();
    for (int gi = 0; gi < NG; ++gi) {
        float s = 0.0f;
#pragma unroll 8
        for (int k = 0; k < FINAL; ++k) s += g[gi * FINAL + k] * w1[k * FINAL + t];
        hdn[gi * FINAL + t] = fmaxf(s + b1[t], 0.0f);
    }
    __syncthreads();
    if (t < NG) {
        float s = 0.0f;
#pragma unroll 8
        for (int k = 0; k < FINAL; ++k) s += hdn[t * FINAL + k] * w2[k];
        out[t] = s + b2[0];
    }
}

// ---------------------------------------------------------------------------
extern "C" void kernel_launch(void* const* d_in, const int* in_sizes, int n_in,
                              void* d_out, int out_size, void* d_ws, size_t ws_size,
                              hipStream_t stream)
{
    const float* x     = (const float*)d_in[0];
    const int*   e0    = (const int*)d_in[1];
    const int*   e1    = (const int*)d_in[2];
    const int*   e2    = (const int*)d_in[3];
    const int*   batch = (const int*)d_in[4];
    const float* W_emb = (const float*)d_in[5];
    const float* qkv   = (const float*)d_in[6];
    const float* fw1   = (const float*)d_in[7];
    const float* fb1   = (const float*)d_in[8];
    const float* fw2   = (const float*)d_in[9];
    const float* fb2   = (const float*)d_in[10];
    const float* mw1   = (const float*)d_in[11];
    const float* mb1   = (const float*)d_in[12];
    const float* mw2   = (const float*)d_in[13];
    const float* mb2   = (const float*)d_in[14];

    int N = in_sizes[0] / 112;
    int E = in_sizes[1] / 2;

    size_t nf   = (size_t)N * FINAL;
    float* h    = (float*)d_ws;
    float* xacc = h + nf;
    float* tmp  = xacc + nf;
    float* sums = tmp + nf;
    float* cnts = sums + NG * FINAL;

    encode_kernel<<<N, 128, 0, stream>>>(x, W_emb, h, N);

    const int* esrc[3] = {e0, e1, e2};
    int fa_grid = (N + 15) / 16;
    int sc_grid = (int)(((size_t)E * 32 + 255) / 256);
    int r_grid  = (int)((nf / 4 + 255) / 256);

    for (int l = 0; l < 2; ++l) {
        int m0 = l * 4;
        featatt_kernel<<<fa_grid, 128, 0, stream>>>(
            h, qkv + (size_t)m0 * 32 * 96, fw1 + (size_t)m0 * 1024, fb1 + m0 * 32,
            fw2 + (size_t)m0 * 1024, fb2 + m0 * 32, xacc, N);
        for (int e = 0; e < 3; ++e) {
            int m = m0 + 1 + e;
            featatt_kernel<<<fa_grid, 128, 0, stream>>>(
                h, qkv + (size_t)m * 32 * 96, fw1 + (size_t)m * 1024, fb1 + m * 32,
                fw2 + (size_t)m * 1024, fb2 + m * 32, tmp, N);
            scatter_kernel<<<sc_grid, 256, 0, stream>>>(tmp, esrc[e], esrc[e] + E, E, xacc);
        }
        relu_kernel<<<r_grid, 256, 0, stream>>>((const float4*)xacc, (float4*)h, (int)(nf / 4));
    }

    hipMemsetAsync(sums, 0, (NG * FINAL + NG) * sizeof(float), stream);
    pool_kernel<<<(N + 255) / 256, 128, 0, stream>>>(h, batch, N, sums, cnts);
    final_kernel<<<1, 128, 0, stream>>>(sums, cnts, mw1, mb1, mw2, mb2, (float*)d_out);
}
